// VGAE_57526791962638
// MI455X (gfx1250) — matrix operand order repaired and checked by reading, not compiled, not run on hardware
//
#include <hip/hip_runtime.h>
#include <hip/hip_bf16.h>
#include <math.h>

#define DI __device__ __forceinline__

typedef __bf16 bf16_t;
typedef __attribute__((ext_vector_type(16))) __bf16 v16bf;
typedef __attribute__((ext_vector_type(8)))  __bf16 v8bf;
typedef __attribute__((ext_vector_type(8)))  float  v8f;

static constexpr int NNODE = 8192;
static constexpr int NEDGE = 262144;
static constexpr int F_IN  = 512;
static constexpr int HDIM  = 256;
static constexpr int LDIM  = 64;

// ---------------- helpers ----------------

DI bf16_t f32_to_bf16(float f) {
  unsigned int u = __builtin_bit_cast(unsigned int, f);
  u += 0x7FFFu + ((u >> 16) & 1u);            // round-to-nearest-even
  unsigned short h = (unsigned short)(u >> 16);
  return __builtin_bit_cast(bf16_t, h);
}

// Fast sigmoid: avoid the IEEE v_div_scale/v_div_fmas expansion.
// Prefer gfx1250's v_tanh_f32 (1 trans op): sigmoid(x) = 0.5*tanh(x/2)+0.5.
// Fallback: v_exp_f32 + v_rcp_f32 via amdgcn builtins.
DI float fast_sigmoid(float x) {
#if __has_builtin(__builtin_amdgcn_tanhf)
  return __builtin_amdgcn_tanhf(0.5f * x) * 0.5f + 0.5f;
#elif __has_builtin(__builtin_amdgcn_exp2f) && __has_builtin(__builtin_amdgcn_rcpf)
  float e = __builtin_amdgcn_exp2f(x * -1.44269504088896341f);   // exp(-x)
  return __builtin_amdgcn_rcpf(1.0f + e);
#else
  return 1.0f / (1.0f + __expf(-x));
#endif
}

// A-matrix (16x32 bf16, MxK) per-lane fragment, per CDNA5 ISA layout:
//   lane L: M = L&15, g = L>>4
//   elements 0..7  <-> K = k0 + 8g + e
//   elements 8..15 <-> K = k0 + 16 + 8g + (e-8)
DI v16bf load_a16(const bf16_t* __restrict__ A, int lda, int row0, int k0, int lane) {
  int m = row0 + (lane & 15);
  int g = lane >> 4;
  const bf16_t* p = A + (size_t)m * lda + k0 + 8 * g;
  v8bf lo = *(const v8bf*)(p);
  v8bf hi = *(const v8bf*)(p + 16);
  return __builtin_shufflevector(lo, hi, 0, 1, 2, 3, 4, 5, 6, 7,
                                 8, 9, 10, 11, 12, 13, 14, 15);
}

// B-matrix (32x16 bf16, KxN) fragment, B given transposed (Bt[n][k], row-major NxK):
//   lane L: N = L&15, g = L>>4; elements 0..7 <-> K = k0 + 8g + e, elements 8..15 <-> K = k0 + 16 + 8g + (e-8)
DI v16bf load_bt16(const bf16_t* __restrict__ Bt, int ldb, int col0, int k0, int lane) {
  int n = col0 + (lane & 15);
  int g = lane >> 4;
  const bf16_t* p = Bt + (size_t)n * ldb + k0 + 8 * g;
  v8bf lo = *(const v8bf*)(p);
  v8bf hi = *(const v8bf*)(p + 16);
  return __builtin_shufflevector(lo, hi, 0, 1, 2, 3, 4, 5, 6, 7,
                                 8, 9, 10, 11, 12, 13, 14, 15);
}

// ---------------- kernels ----------------

__global__ void cvt_f32_to_bf16(const float* __restrict__ in, bf16_t* __restrict__ out, int n) {
  int i = blockIdx.x * blockDim.x + threadIdx.x;
  if (i < n) out[i] = f32_to_bf16(in[i]);
}

__global__ void fill_zero_f4(float4* __restrict__ p, int n4) {
  int i = blockIdx.x * blockDim.x + threadIdx.x;
  if (i < n4) p[i] = make_float4(0.f, 0.f, 0.f, 0.f);
}

// out[dst[e]][:] += x[src[e]][:]  (float4-vectorized, cols4 = 1<<c4shift)
__global__ void scatter_add_f4(const float4* __restrict__ x, const int* __restrict__ src,
                               const int* __restrict__ dst, float* __restrict__ out,
                               int c4shift, int total) {
  int i = blockIdx.x * blockDim.x + threadIdx.x;
  if (i >= total) return;
  int c = i & ((1 << c4shift) - 1);
  int e = i >> c4shift;
  int s = src[e];
  int d = dst[e];
  float4 v = x[((size_t)s << c4shift) + c];
  float* o = out + ((((size_t)d << c4shift) + c) << 2);
  atomicAdd(o + 0, v.x);
  atomicAdd(o + 1, v.y);
  atomicAdd(o + 2, v.z);
  atomicAdd(o + 3, v.w);
}

__global__ void bias_relu_to_bf16(const float* __restrict__ agg, const float* __restrict__ bias,
                                  bf16_t* __restrict__ out, int colmask, int n) {
  int i = blockIdx.x * blockDim.x + threadIdx.x;
  if (i >= n) return;
  float v = agg[i] + bias[i & colmask];
  v = v > 0.f ? v : 0.f;
  out[i] = f32_to_bf16(v);
}

__global__ void bias_add_and_bf16(float* __restrict__ io, const float* __restrict__ bias,
                                  bf16_t* __restrict__ outb, int colmask, int n) {
  int i = blockIdx.x * blockDim.x + threadIdx.x;
  if (i >= n) return;
  float v = io[i] + bias[i & colmask];
  io[i] = v;
  if (outb) outb[i] = f32_to_bf16(v);
}

// C[M x NC] = A[M x K] * Bt[NC x K]^T, bf16 inputs, f32 accumulate.
// One wave (32 threads) computes a 64x32 block: 4 M-tiles x 2 N-tiles of 16x16,
// 8 v_wmma_f32_16x16x32_bf16 per 32-wide k-step. NC compile-time so epilogue
// stores use one base pointer per tile + immediate row offsets (r*NC*4 < 2^23).
template <bool SIG, int NC>
__global__ __launch_bounds__(32) void gemm_wmma_bf16f32(
    const bf16_t* __restrict__ A, const bf16_t* __restrict__ Bt,
    float* __restrict__ C, int K) {
  const int lane = threadIdx.x;
  const int n0 = blockIdx.x * 32;
  const int m0 = blockIdx.y * 64;

  v8f acc[4][2] = {};

  for (int k0 = 0; k0 < K; k0 += 32) {
    v16bf a[4];
    v16bf b[2];
#pragma unroll
    for (int mt = 0; mt < 4; ++mt) a[mt] = load_a16(A, K, m0 + 16 * mt, k0, lane);
#pragma unroll
    for (int nt = 0; nt < 2; ++nt) b[nt] = load_bt16(Bt, K, n0 + 16 * nt, k0, lane);
#pragma unroll
    for (int mt = 0; mt < 4; ++mt) {
#pragma unroll
      for (int nt = 0; nt < 2; ++nt) {
        acc[mt][nt] = __builtin_amdgcn_wmma_f32_16x16x32_bf16(
            false, a[mt], false, b[nt], (short)0, acc[mt][nt], false, false);
      }
    }
  }

  // C/D fragment layout: row = r + 8*(lane>>4), col = lane&15 (per 16x16 tile)
  const int g = lane >> 4;
  const int nl = lane & 15;
#pragma unroll
  for (int mt = 0; mt < 4; ++mt) {
#pragma unroll
    for (int nt = 0; nt < 2; ++nt) {
      float* cptr = C + (size_t)(m0 + 16 * mt + 8 * g) * NC + (n0 + 16 * nt + nl);
#pragma unroll
      for (int r = 0; r < 8; ++r) {
        float v = acc[mt][nt][r];
        if (SIG) v = fast_sigmoid(v);
        cptr[(size_t)r * NC] = v;   // NC compile-time -> immediate offsets
      }
    }
  }
}

// ---------------- host launcher ----------------

extern "C" void kernel_launch(void* const* d_in, const int* in_sizes, int n_in,
                              void* d_out, int out_size, void* d_ws, size_t ws_size,
                              hipStream_t stream) {
  (void)in_sizes; (void)n_in; (void)out_size; (void)ws_size;

  const float* features = (const float*)d_in[0];
  const int*   src      = (const int*)d_in[1];
  const int*   dst      = (const int*)d_in[2];
  const float* W1       = (const float*)d_in[3];
  const float* b1       = (const float*)d_in[4];
  const float* W2       = (const float*)d_in[5];
  const float* b2       = (const float*)d_in[6];
  const float* W3       = (const float*)d_in[7];
  const float* b3       = (const float*)d_in[8];

  float* out    = (float*)d_out;
  float* adj    = out;                                  // [N, N]
  float* mu     = out + (size_t)NNODE * NNODE;          // [N, L]
  float* logvar = mu + (size_t)NNODE * LDIM;            // [N, L]

  // workspace carve-out (~35 MB), 256B aligned slabs
  char* ws = (char*)d_ws;
  size_t off = 0;
  auto carve = [&](size_t bytes) -> void* {
    void* p = ws + off;
    off = (off + bytes + 255) & ~(size_t)255;
    return p;
  };
  bf16_t* Fb      = (bf16_t*)carve((size_t)NNODE * F_IN * sizeof(bf16_t));
  bf16_t* W1b     = (bf16_t*)carve((size_t)HDIM * F_IN * sizeof(bf16_t));
  bf16_t* W2b     = (bf16_t*)carve((size_t)LDIM * HDIM * sizeof(bf16_t));
  bf16_t* W3b     = (bf16_t*)carve((size_t)LDIM * HDIM * sizeof(bf16_t));
  float*  h0      = (float*)carve((size_t)NNODE * HDIM * sizeof(float));
  float*  aggH    = (float*)carve((size_t)NNODE * HDIM * sizeof(float));
  bf16_t* hiddenb = (bf16_t*)carve((size_t)NNODE * HDIM * sizeof(bf16_t));
  float*  p2      = (float*)carve((size_t)NNODE * LDIM * sizeof(float));
  float*  p3      = (float*)carve((size_t)NNODE * LDIM * sizeof(float));
  bf16_t* mub     = (bf16_t*)carve((size_t)NNODE * LDIM * sizeof(bf16_t));

  const int T = 256;
  auto blocks = [&](int n) { return (n + T - 1) / T; };

  // bf16 copies of features and weights (weights stored [out,in] == Bt layout)
  cvt_f32_to_bf16<<<blocks(NNODE * F_IN), T, 0, stream>>>(features, Fb, NNODE * F_IN);
  cvt_f32_to_bf16<<<blocks(HDIM * F_IN), T, 0, stream>>>(W1, W1b, HDIM * F_IN);
  cvt_f32_to_bf16<<<blocks(LDIM * HDIM), T, 0, stream>>>(W2, W2b, LDIM * HDIM);
  cvt_f32_to_bf16<<<blocks(LDIM * HDIM), T, 0, stream>>>(W3, W3b, LDIM * HDIM);

  // h0 = features @ W1^T   [N, H]   (linearity: agg(x) @ W^T == agg(x @ W^T))
  gemm_wmma_bf16f32<false, HDIM><<<dim3(HDIM / 32, NNODE / 64), 32, 0, stream>>>(
      Fb, W1b, h0, F_IN);

  // aggH[dst] += h0[src] ; hidden = relu(aggH + b1) -> bf16
  fill_zero_f4<<<blocks(NNODE * HDIM / 4), T, 0, stream>>>((float4*)aggH, NNODE * HDIM / 4);
  scatter_add_f4<<<blocks(NEDGE * (HDIM / 4)), T, 0, stream>>>(
      (const float4*)h0, src, dst, aggH, 6 /* 64 float4 cols */, NEDGE * (HDIM / 4));
  bias_relu_to_bf16<<<blocks(NNODE * HDIM), T, 0, stream>>>(
      aggH, b1, hiddenb, HDIM - 1, NNODE * HDIM);

  // p2 = hidden @ W2^T ; p3 = hidden @ W3^T   [N, L]
  gemm_wmma_bf16f32<false, LDIM><<<dim3(LDIM / 32, NNODE / 64), 32, 0, stream>>>(
      hiddenb, W2b, p2, HDIM);
  gemm_wmma_bf16f32<false, LDIM><<<dim3(LDIM / 32, NNODE / 64), 32, 0, stream>>>(
      hiddenb, W3b, p3, HDIM);

  // mu = agg(p2) + b2 ; logvar = agg(p3) + b3   (accumulate directly in d_out)
  fill_zero_f4<<<blocks(NNODE * LDIM / 4), T, 0, stream>>>((float4*)mu, NNODE * LDIM / 4);
  fill_zero_f4<<<blocks(NNODE * LDIM / 4), T, 0, stream>>>((float4*)logvar, NNODE * LDIM / 4);
  scatter_add_f4<<<blocks(NEDGE * (LDIM / 4)), T, 0, stream>>>(
      (const float4*)p2, src, dst, mu, 4 /* 16 float4 cols */, NEDGE * (LDIM / 4));
  scatter_add_f4<<<blocks(NEDGE * (LDIM / 4)), T, 0, stream>>>(
      (const float4*)p3, src, dst, logvar, 4, NEDGE * (LDIM / 4));
  bias_add_and_bf16<<<blocks(NNODE * LDIM), T, 0, stream>>>(
      mu, b2, mub, LDIM - 1, NNODE * LDIM);
  bias_add_and_bf16<<<blocks(NNODE * LDIM), T, 0, stream>>>(
      logvar, b3, (bf16_t*)nullptr, LDIM - 1, NNODE * LDIM);

  // adj = sigmoid(mu @ mu^T)   [N, N] — HBM-store-bound (~256 MB)
  gemm_wmma_bf16f32<true, NNODE><<<dim3(NNODE / 32, NNODE / 64), 32, 0, stream>>>(
      mub, mub, adj, LDIM);
}
